// FMHA_79448305041613
// MI455X (gfx1250) — matrix operand order repaired and checked by reading, not compiled, not run on hardware
//
#include <hip/hip_runtime.h>

typedef __attribute__((ext_vector_type(16))) _Float16 v16h;
typedef __attribute__((ext_vector_type(8)))  _Float16 v8h;
typedef __attribute__((ext_vector_type(4)))  _Float16 v4h;
typedef __attribute__((ext_vector_type(8)))  float    v8f;

#define N_HEADS 16
#define HDIM    64
#define HID_SZ  (N_HEADS*HDIM)   // 1024
#define Q_TILE  128
#define K_TILE  64
#define THREADS 256

union V16U { v16h v; v8h h[2]; };

__global__ __launch_bounds__(THREADS, 1)
void fmha_fwd_wmma(const float* __restrict__ qkv, const int* __restrict__ cu,
                   float* __restrict__ out)
{
    // LDS: 16 + 8 + 8 + 16 = 48 KB (of 320 KB/WGP)
    __shared__ __align__(32) _Float16 sQ [Q_TILE*HDIM];   // [qrow][d], f16, pre-scaled
    __shared__ __align__(32) _Float16 sK [K_TILE*HDIM];   // [key][d]
    __shared__ __align__(32) _Float16 sVT[HDIM*K_TILE];   // [d][key]  (transposed V)
    __shared__ __align__(32) _Float16 sP [8*16*K_TILE];   // per-wave P^T scratch [q][key]

    const int b    = blockIdx.z;
    const int h    = blockIdx.y;
    const int qt   = blockIdx.x;
    const int tid  = threadIdx.x;
    const int wave = tid >> 5;
    const int lane = tid & 31;
    const int lmod = lane & 15;
    const int lhi  = lane >> 4;          // 0 or 1

    const int seq_start = cu[b];
    const int seq_len   = cu[b+1] - seq_start;
    const int q0        = qt * Q_TILE;
    if (q0 >= seq_len) return;

    // ---------------- stage Q tile (scaled by 1/sqrt(D)) ----------------
    {
        const float scale = 0.125f;
        int row  = tid >> 1;
        int seg  = (tid & 1) * 32;
        int qrow = q0 + row;
        _Float16* dst = &sQ[row*HDIM + seg];
        if (qrow < seq_len) {
            const float* src = qkv + (size_t)(seq_start + qrow)*(3*HID_SZ) + h*HDIM + seg;
            const float4* s4 = (const float4*)src;
            #pragma unroll
            for (int j = 0; j < 8; ++j) {
                float4 f = s4[j];
                v4h p = { (_Float16)(f.x*scale), (_Float16)(f.y*scale),
                          (_Float16)(f.z*scale), (_Float16)(f.w*scale) };
                *(v4h*)(dst + 4*j) = p;
            }
        } else {
            v4h z = { (_Float16)0.f, (_Float16)0.f, (_Float16)0.f, (_Float16)0.f };
            #pragma unroll
            for (int j = 0; j < 8; ++j) *(v4h*)(dst + 4*j) = z;
        }
    }
    __syncthreads();

    // -------- per-wave Q^T B-fragments (this wave's 16 query columns) -----
    // B layout (32x16 f16): lane owns column N=lmod (query); lanes 0-15 hold
    // K(=d) 0..15 of the chunk, lanes 16-31 hold K 16..31 -> contiguous v16h.
    v16h qbfrag[2];
    #pragma unroll
    for (int c = 0; c < 2; ++c)
        qbfrag[c] = *(const v16h*)&sQ[(wave*16 + lmod)*HDIM + c*32 + lhi*16];

    // O^T accumulators: tile t covers d rows 16t..16t+15; lane owns query lmod.
    v8f   oacc[4] = {};
    float mstat = -1e30f, lstat = 0.f;

    _Float16* sPw = &sP[wave * (16*K_TILE)];

    const int nkb = (seq_len + K_TILE - 1) / K_TILE;
    for (int kb = 0; kb < nkb; ++kb) {
        __syncthreads();
        // ------------ stage K (row-major) and V (transposed), 64 keys -------
        #pragma unroll
        for (int kk = 0; kk < 2; ++kk) {
            int key  = kk*32 + (tid >> 3);  // 0..63
            int seg  = (tid & 7) * 8;       // d base, 0..56
            int krow = kb*K_TILE + key;
            if (krow < seq_len) {
                const float* kp = qkv + (size_t)(seq_start + krow)*(3*HID_SZ)
                                      + HID_SZ + h*HDIM + seg;
                const float* vp = kp + HID_SZ;
                float4 k0 = ((const float4*)kp)[0];
                float4 k1 = ((const float4*)kp)[1];
                v8h kh = { (_Float16)k0.x, (_Float16)k0.y, (_Float16)k0.z, (_Float16)k0.w,
                           (_Float16)k1.x, (_Float16)k1.y, (_Float16)k1.z, (_Float16)k1.w };
                *(v8h*)&sK[key*HDIM + seg] = kh;
                float4 u0 = ((const float4*)vp)[0];
                float4 u1 = ((const float4*)vp)[1];
                sVT[(seg+0)*K_TILE + key] = (_Float16)u0.x;
                sVT[(seg+1)*K_TILE + key] = (_Float16)u0.y;
                sVT[(seg+2)*K_TILE + key] = (_Float16)u0.z;
                sVT[(seg+3)*K_TILE + key] = (_Float16)u0.w;
                sVT[(seg+4)*K_TILE + key] = (_Float16)u1.x;
                sVT[(seg+5)*K_TILE + key] = (_Float16)u1.y;
                sVT[(seg+6)*K_TILE + key] = (_Float16)u1.z;
                sVT[(seg+7)*K_TILE + key] = (_Float16)u1.w;
                if (krow + K_TILE < seq_len)
                    __builtin_prefetch(kp + (size_t)K_TILE*3*HID_SZ, 0, 1);
            } else {
                v8h z8 = { (_Float16)0.f,(_Float16)0.f,(_Float16)0.f,(_Float16)0.f,
                           (_Float16)0.f,(_Float16)0.f,(_Float16)0.f,(_Float16)0.f };
                *(v8h*)&sK[key*HDIM + seg] = z8;
                #pragma unroll
                for (int j = 0; j < 8; ++j) sVT[(seg+j)*K_TILE + key] = (_Float16)0.f;
            }
        }
        __syncthreads();

        // -------- S^T(64x16) = K(64x64) Q^T(64x16): rows=keys, cols=queries
        // A frag (16x32): lane row = key 16t+lmod; d-chunk split lo/hi by lhi.
        v8f sc[4];
        #pragma unroll
        for (int t = 0; t < 4; ++t) {
            v8f c = {};
            #pragma unroll
            for (int cc = 0; cc < 2; ++cc) {
                const _Float16* base = &sK[(t*16 + lmod)*HDIM + cc*32];
                V16U u;
                u.h[0] = *(const v8h*)(base + lhi*16);
                u.h[1] = *(const v8h*)(base + lhi*16 + 8);
                c = __builtin_amdgcn_wmma_f32_16x16x32_f16(false, u.v, false, qbfrag[cc],
                                                           (short)0, c, false, false);
            }
            sc[t] = c;
        }

        // -------- mask tail keys (rows of S^T): key = kb*64 + 16t + 8*lhi + r
        if (kb*K_TILE + K_TILE > seq_len) {
            #pragma unroll
            for (int t = 0; t < 4; ++t) {
                #pragma unroll
                for (int r = 0; r < 8; ++r) {
                    int keyidx = kb*K_TILE + t*16 + 8*lhi + r;
                    if (keyidx >= seq_len) sc[t][r] = -1e9f;
                }
            }
        }

        // -------- online softmax: lane owns one query column ----------------
        float x = sc[0][0];
        #pragma unroll
        for (int t = 0; t < 4; ++t)
            #pragma unroll
            for (int r = 0; r < 8; ++r) x = fmaxf(x, sc[t][r]);
        x = fmaxf(x, __shfl_xor(x, 16, 32));       // combine the other 32 keys

        float mn    = fmaxf(mstat, x);
        float alpha = __expf(mstat - mn);
        mstat = mn;

        // exp + pack P^T [query][key]; consecutive r = consecutive keys
        float psum = 0.f;
        #pragma unroll
        for (int t = 0; t < 4; ++t) {
            float p[8];
            #pragma unroll
            for (int r = 0; r < 8; ++r) { p[r] = __expf(sc[t][r] - mn); psum += p[r]; }
            v8h ph = { (_Float16)p[0], (_Float16)p[1], (_Float16)p[2], (_Float16)p[3],
                       (_Float16)p[4], (_Float16)p[5], (_Float16)p[6], (_Float16)p[7] };
            *(v8h*)&sPw[lmod*K_TILE + t*16 + 8*lhi] = ph;   // keys 16t+8*lhi..+7
        }
        psum += __shfl_xor(psum, 16, 32);
        lstat = lstat*alpha + psum;

        #pragma unroll
        for (int t = 0; t < 4; ++t)
            #pragma unroll
            for (int r = 0; r < 8; ++r) oacc[t][r] *= alpha;

        // -------- O^T(64x16) += V^T(64x64) P^T(64x16) -----------------------
        v16h pfrag[2];
        #pragma unroll
        for (int cc = 0; cc < 2; ++cc)
            pfrag[cc] = *(const v16h*)&sPw[lmod*K_TILE + cc*32 + lhi*16];
        #pragma unroll
        for (int t = 0; t < 4; ++t) {
            #pragma unroll
            for (int cc = 0; cc < 2; ++cc) {
                const _Float16* vb = &sVT[(t*16 + lmod)*K_TILE + cc*32]; // A: row d
                V16U u;
                u.h[0] = *(const v8h*)(vb + lhi*16);
                u.h[1] = *(const v8h*)(vb + lhi*16 + 8);
                oacc[t] = __builtin_amdgcn_wmma_f32_16x16x32_f16(false, u.v, false, pfrag[cc],
                                                                 (short)0, oacc[t], false, false);
            }
        }
    }

    // ---------------- normalize and store (lane owns one query row) --------
    {
        int q = q0 + wave*16 + lmod;
        if (q < seq_len) {
            float rcp = lstat > 0.f ? 1.f / lstat : 0.f;
            float* op = out + (size_t)(seq_start + q)*HID_SZ + h*HDIM;
            #pragma unroll
            for (int t = 0; t < 4; ++t) {
                int dbase = t*16 + 8*lhi;               // O^T row d -> out column d
                float4 a = { oacc[t][0]*rcp, oacc[t][1]*rcp, oacc[t][2]*rcp, oacc[t][3]*rcp };
                float4 c = { oacc[t][4]*rcp, oacc[t][5]*rcp, oacc[t][6]*rcp, oacc[t][7]*rcp };
                *(float4*)(op + dbase)     = a;
                *(float4*)(op + dbase + 4) = c;
            }
        }
    }
}

extern "C" void kernel_launch(void* const* d_in, const int* in_sizes, int n_in,
                              void* d_out, int out_size, void* d_ws, size_t ws_size,
                              hipStream_t stream) {
    (void)n_in; (void)d_ws; (void)ws_size;
    const float* qkv = (const float*)d_in[0];
    const int*   cu  = (const int*)d_in[1];
    float*       out = (float*)d_out;

    int B      = in_sizes[1] - 1;                // cu_seqlens has B+1 entries
    int total  = out_size / HID_SZ;              // total tokens
    int qtiles = (total + Q_TILE - 1) / Q_TILE;  // conservative per-batch bound

    dim3 grid(qtiles, N_HEADS, B);
    dim3 block(THREADS);
    fmha_fwd_wmma<<<grid, block, 0, stream>>>(qkv, cu, out);
}
